// CausalAttentionBlock_5188320493936
// MI455X (gfx1250) — matrix operand order repaired and checked by reading, not compiled, not run on hardware
//
#include <hip/hip_runtime.h>
#include <hip/hip_bf16.h>
#include <math.h>

// ---------------------------------------------------------------------------
// Problem constants (from reference): B=2, L=2048, D=1024, H=16, FF=4096, HID=256
// ---------------------------------------------------------------------------
#define CB   2
#define CL   2048
#define CD   1024
#define CH   16
#define CE   64        // D / H
#define CFF  4096
#define CHID 256
#define CBL  (CB * CL)               // 4096
#define CBLD (CB * CL * CD)          // 4,194,304

typedef __attribute__((ext_vector_type(16))) __bf16 v16bf;
typedef __attribute__((ext_vector_type(8)))  float  v8f;

__device__ __forceinline__ float gelu_f(float x) {
  // exact erf GELU (torch nn.GELU default)
  return 0.5f * x * (1.0f + erff(x * 0.70710678118654752f));
}
__device__ __forceinline__ float sigmoid_f(float x) {
  return 1.0f / (1.0f + __expf(-x));
}
__device__ __forceinline__ void store_bf16x4(__bf16* p, float4 v) {
  p[0] = (__bf16)v.x; p[1] = (__bf16)v.y; p[2] = (__bf16)v.z; p[3] = (__bf16)v.w;
}

// ---------------------------------------------------------------------------
// Generic GEMM: C[M,N] = act(A[M,K] @ W[N,K]^T + bias[N])
// Block 256 threads = 8 waves. Block tile 128x128; wave tile 32x64:
//   2 A-frags x 4 B-frags -> 8 v_wmma_f32_16x16x32_bf16 per k-step per wave.
// Software-pipelined: next k-tile's global float4 loads are issued before the
// WMMA section, consumed (cvt->bf16 LDS store) one iteration later.
// Requires: N % 128 == 0, K % 32 == 0. M may be ragged (bounds-checked).
// ---------------------------------------------------------------------------
__global__ __launch_bounds__(256) void gemm_bias_kernel(
    const float* __restrict__ A, const float* __restrict__ W,
    const float* __restrict__ bias, float* __restrict__ C,
    int M, int N, int K, int act)
{
  __shared__ __bf16 As[128 * 32];
  __shared__ __bf16 Ws[128 * 32];

  const int tid  = threadIdx.x;
  const int lane = tid & 31;
  const int w    = tid >> 5;
  const int wm   = w & 3;        // 4 m-subtiles of 32 rows
  const int wn   = w >> 2;       // 2 n-subtiles of 64 cols
  const int nl   = lane & 15;
  const int kh   = lane >> 4;
  const int m0   = blockIdx.x * 128;
  const int n0   = blockIdx.y * 128;
  const bool full_tile = (m0 + 128 <= M);   // fast path: no per-lane M guard

  float4 ar[4], wr[4];                       // pipeline registers
  auto load_tile = [&](int k0) {
    #pragma unroll
    for (int j = 0; j < 4; ++j) {
      int idx = tid * 4 + j * 1024;          // 4096 elements per tile
      int r = idx >> 5, c = idx & 31;
      int gr = m0 + r;
      if (full_tile || gr < M) ar[j] = *(const float4*)(A + (size_t)gr * K + k0 + c);
      else                     ar[j] = make_float4(0.f, 0.f, 0.f, 0.f);
      wr[j] = *(const float4*)(W + (size_t)(n0 + r) * K + k0 + c);
    }
  };

  v8f acc[2][4] = {};
  load_tile(0);

  for (int k0 = 0; k0 < K; k0 += 32) {
    __syncthreads();   // prior iteration's LDS reads complete
    #pragma unroll
    for (int j = 0; j < 4; ++j) {
      int idx = tid * 4 + j * 1024;
      int r = idx >> 5, c = idx & 31;
      store_bf16x4(&As[r * 32 + c], ar[j]);
      store_bf16x4(&Ws[r * 32 + c], wr[j]);
    }
    __syncthreads();
    if (k0 + 32 < K) load_tile(k0 + 32);    // prefetch next tile behind WMMAs

    // A fragments: two 16x32 tiles (contiguous k = 16*kh + s per lane half)
    v16bf af[2];
    #pragma unroll
    for (int t = 0; t < 2; ++t) {
      const __bf16* ap = &As[(wm * 32 + t * 16 + nl) * 32];
      #pragma unroll
      for (int v = 0; v < 8; ++v) {
        int k = kh * 16 + 2 * v;
        af[t][2 * v]     = ap[k];
        af[t][2 * v + 1] = ap[k + 1];
      }
    }
    // B fragments: four 32x16 tiles, B[k][n] = W[n][k] (contiguous k = 16*kh + s, same as A)
    v16bf bf[4];
    #pragma unroll
    for (int t = 0; t < 4; ++t) {
      const __bf16* wp = &Ws[(wn * 64 + t * 16 + nl) * 32];
      #pragma unroll
      for (int v = 0; v < 8; ++v) {
        int k = kh * 16 + 2 * v;
        bf[t][2 * v]     = wp[k];
        bf[t][2 * v + 1] = wp[k + 1];
      }
    }
    #pragma unroll
    for (int i = 0; i < 2; ++i)
      #pragma unroll
      for (int jn = 0; jn < 4; ++jn)
        acc[i][jn] = __builtin_amdgcn_wmma_f32_16x16x32_bf16(
            false, af[i], false, bf[jn], (short)0, acc[i][jn], false, false);
  }

  // Epilogue: bias + optional GELU, bounds-checked on M
  #pragma unroll
  for (int i = 0; i < 2; ++i) {
    #pragma unroll
    for (int r = 0; r < 8; ++r) {
      int row = m0 + wm * 32 + i * 16 + r + 8 * kh;
      if (row >= M) continue;
      #pragma unroll
      for (int jn = 0; jn < 4; ++jn) {
        int col = n0 + wn * 64 + jn * 16 + nl;
        float vv = acc[i][jn][r] + bias[col];
        if (act) vv = gelu_f(vv);
        C[(size_t)row * N + col] = vv;
      }
    }
  }
}

// ---------------------------------------------------------------------------
// Flash attention over one (batch, head): softmax((Q K^T)/8 [*1/tau + te/tau]) V
// Q/K/V stored as [B, L, D] fp32 (head h occupies cols h*64 .. h*64+63).
// Block = 128 threads (4 waves); each wave owns 16 query rows, kv tiles of 32.
// K/V staging is software-pipelined one kv-tile ahead.
// Grid: (L/64, H, B). te_v/tau_v nullptr => self-attention (scale 1/8, bias 0).
// ---------------------------------------------------------------------------
__global__ __launch_bounds__(128) void attn_kernel(
    const float* __restrict__ Qg, const float* __restrict__ Kg,
    const float* __restrict__ Vg, float* __restrict__ Og,
    const float* __restrict__ te_v, const float* __restrict__ tau_v,
    int Lq, int S)
{
  __shared__ __bf16 Qs[64 * 64];
  __shared__ __bf16 Ks[32 * 64];
  __shared__ __bf16 Vs[32 * 64];
  __shared__ __bf16 Ps[4][16 * 32];

  const int tid  = threadIdx.x;
  const int lane = tid & 31;
  const int w    = tid >> 5;
  const int nl   = lane & 15;
  const int kh   = lane >> 4;
  const int q0   = blockIdx.x * 64;
  const int h    = blockIdx.y;
  const int b    = blockIdx.z;

  // Stage 64x64 Q tile for this block (float4 loads)
  const size_t baseQ = ((size_t)b * Lq + q0) * CD + h * CE;
  #pragma unroll
  for (int j = 0; j < 8; ++j) {
    int idx = tid * 4 + j * 512;             // 4096 elements
    int r = idx >> 6, c = idx & 63;
    float4 qv = *(const float4*)(Qg + baseQ + (size_t)r * CD + c);
    store_bf16x4(&Qs[r * 64 + c], qv);
  }

  // Pipeline registers for K/V staging
  float4 kr[4], vr[4];
  auto load_kv = [&](int s0) {
    const size_t baseK = ((size_t)b * S + s0) * CD + h * CE;
    #pragma unroll
    for (int j = 0; j < 4; ++j) {
      int idx = tid * 4 + j * 512;           // 2048 elements per tile
      int r = idx >> 6, c = idx & 63;
      kr[j] = *(const float4*)(Kg + baseK + (size_t)r * CD + c);
      vr[j] = *(const float4*)(Vg + baseK + (size_t)r * CD + c);
    }
  };
  load_kv(0);
  __syncthreads();

  // Each wave's Q fragments (E = 64 -> two 16x32 A-frags)
  v16bf aq0, aq1;
  {
    const __bf16* qp = &Qs[(w * 16 + nl) * 64];
    #pragma unroll
    for (int v = 0; v < 8; ++v) {
      int k = kh * 16 + 2 * v;
      aq0[2 * v]     = qp[k];
      aq0[2 * v + 1] = qp[k + 1];
      aq1[2 * v]     = qp[32 + k];
      aq1[2 * v + 1] = qp[32 + k + 1];
    }
  }

  float sc = 0.125f, sb = 0.0f;   // 1/sqrt(E) = 1/8
  if (tau_v) { float t = tau_v[b]; sc = 0.125f / t; sb = te_v[b] / t; }

  v8f o0 = {}, o1 = {}, o2 = {}, o3 = {};
  float rmax[8], rsum[8];
  #pragma unroll
  for (int r = 0; r < 8; ++r) { rmax[r] = -3.0e38f; rsum[r] = 0.0f; }

  for (int s0 = 0; s0 < S; s0 += 32) {
    __syncthreads();   // prior iteration's reads of Ks/Vs/Ps done
    #pragma unroll
    for (int j = 0; j < 4; ++j) {
      int idx = tid * 4 + j * 512;
      int r = idx >> 6, c = idx & 63;
      store_bf16x4(&Ks[r * 64 + c], kr[j]);
      store_bf16x4(&Vs[r * 64 + c], vr[j]);
    }
    __syncthreads();
    if (s0 + 32 < S) load_kv(s0 + 32);      // prefetch next kv tile

    // Scores: two 16x16 tiles (kv cols 0..15 and 16..31), K-dim = E in 2 steps
    v8f c0 = {}, c1 = {};
    #pragma unroll
    for (int nt = 0; nt < 2; ++nt) {
      v16bf bk0, bk1;   // B[k=e][n=kv] = K[kvrow][e]
      const __bf16* kp = &Ks[(nt * 16 + nl) * 64];
      #pragma unroll
      for (int v = 0; v < 8; ++v) {
        int k = kh * 16 + 2 * v;
        bk0[2 * v]     = kp[k];
        bk0[2 * v + 1] = kp[k + 1];
        bk1[2 * v]     = kp[32 + k];
        bk1[2 * v + 1] = kp[32 + k + 1];
      }
      if (nt == 0) {
        c0 = __builtin_amdgcn_wmma_f32_16x16x32_bf16(false, aq0, false, bk0, (short)0, c0, false, false);
        c0 = __builtin_amdgcn_wmma_f32_16x16x32_bf16(false, aq1, false, bk1, (short)0, c0, false, false);
      } else {
        c1 = __builtin_amdgcn_wmma_f32_16x16x32_bf16(false, aq0, false, bk0, (short)0, c1, false, false);
        c1 = __builtin_amdgcn_wmma_f32_16x16x32_bf16(false, aq1, false, bk1, (short)0, c1, false, false);
      }
    }

    // Online softmax. C layout: VGPR r -> row (r + 8*kh), col nl (c1: col nl+16).
    #pragma unroll
    for (int r = 0; r < 8; ++r) {
      float s0v = c0[r] * sc + sb;
      float s1v = c1[r] * sc + sb;
      float mx = fmaxf(s0v, s1v);
      mx = fmaxf(mx, __shfl_xor(mx, 1));
      mx = fmaxf(mx, __shfl_xor(mx, 2));
      mx = fmaxf(mx, __shfl_xor(mx, 4));
      mx = fmaxf(mx, __shfl_xor(mx, 8));
      float nm   = fmaxf(rmax[r], mx);
      float corr = __expf(rmax[r] - nm);
      rmax[r] = nm;
      float p0 = __expf(s0v - nm);
      float p1 = __expf(s1v - nm);
      float ps = p0 + p1;
      ps += __shfl_xor(ps, 1);
      ps += __shfl_xor(ps, 2);
      ps += __shfl_xor(ps, 4);
      ps += __shfl_xor(ps, 8);
      rsum[r] = rsum[r] * corr + ps;
      o0[r] *= corr; o1[r] *= corr; o2[r] *= corr; o3[r] *= corr;
      Ps[w][(r + 8 * kh) * 32 + nl]      = (__bf16)p0;
      Ps[w][(r + 8 * kh) * 32 + 16 + nl] = (__bf16)p1;
    }
    __syncthreads();   // make per-wave P tile visible in A-layout read below

    // P fragment (16x32 over kv) in A layout
    v16bf ap;
    {
      const __bf16* pp = &Ps[w][nl * 32];
      #pragma unroll
      for (int v = 0; v < 8; ++v) {
        int k = kh * 16 + 2 * v;
        ap[2 * v]     = pp[k];
        ap[2 * v + 1] = pp[k + 1];
      }
    }
    // out += P @ V for 4 e-tiles of 16
    #pragma unroll
    for (int t = 0; t < 4; ++t) {
      v16bf bv;   // B[k=kv][n=e] = V[kv][t*16+n]
      #pragma unroll
      for (int v = 0; v < 8; ++v) {
        int kv = kh * 16 + 2 * v;
        bv[2 * v]     = Vs[kv * 64 + t * 16 + nl];
        bv[2 * v + 1] = Vs[(kv + 1) * 64 + t * 16 + nl];
      }
      if (t == 0)      o0 = __builtin_amdgcn_wmma_f32_16x16x32_bf16(false, ap, false, bv, (short)0, o0, false, false);
      else if (t == 1) o1 = __builtin_amdgcn_wmma_f32_16x16x32_bf16(false, ap, false, bv, (short)0, o1, false, false);
      else if (t == 2) o2 = __builtin_amdgcn_wmma_f32_16x16x32_bf16(false, ap, false, bv, (short)0, o2, false, false);
      else             o3 = __builtin_amdgcn_wmma_f32_16x16x32_bf16(false, ap, false, bv, (short)0, o3, false, false);
    }
  }

  // Epilogue: normalize by row sums, scatter to [B,L,D]
  const size_t baseO = ((size_t)b * Lq + q0 + w * 16) * CD + h * CE;
  #pragma unroll
  for (int r = 0; r < 8; ++r) {
    float inv = 1.0f / rsum[r];
    int rr = r + 8 * kh;
    Og[baseO + (size_t)rr * CD +  0 + nl] = o0[r] * inv;
    Og[baseO + (size_t)rr * CD + 16 + nl] = o1[r] * inv;
    Og[baseO + (size_t)rr * CD + 32 + nl] = o2[r] * inv;
    Og[baseO + (size_t)rr * CD + 48 + nl] = o3[r] * inv;
  }
}

// ---------------------------------------------------------------------------
// y = LayerNorm(res + g * xin), row length D=1024, one block per row.
// gate: per-batch scale (nullptr => 1.0).
// ---------------------------------------------------------------------------
__global__ __launch_bounds__(256) void add_ln_kernel(
    const float* __restrict__ res, const float* __restrict__ xin,
    const float* __restrict__ gate,
    const float* __restrict__ gamma, const float* __restrict__ beta,
    float* __restrict__ out, int Lr)
{
  __shared__ float red[256];
  const int row = blockIdx.x;
  const int bb  = row / Lr;
  const int tid = threadIdx.x;
  const float g = gate ? gate[bb] : 1.0f;
  const float* rp = res + (size_t)row * CD;
  const float* xp = xin + (size_t)row * CD;

  float vals[4];
  float s = 0.0f;
  #pragma unroll
  for (int i = 0; i < 4; ++i) {
    int d = tid + i * 256;
    vals[i] = rp[d] + g * xp[d];
    s += vals[i];
  }
  red[tid] = s; __syncthreads();
  for (int st = 128; st > 0; st >>= 1) { if (tid < st) red[tid] += red[tid + st]; __syncthreads(); }
  float mean = red[0] * (1.0f / CD);
  __syncthreads();
  float vs = 0.0f;
  #pragma unroll
  for (int i = 0; i < 4; ++i) { float d = vals[i] - mean; vs += d * d; }
  red[tid] = vs; __syncthreads();
  for (int st = 128; st > 0; st >>= 1) { if (tid < st) red[tid] += red[tid + st]; __syncthreads(); }
  float rinv = rsqrtf(red[0] * (1.0f / CD) + 1e-5f);
  #pragma unroll
  for (int i = 0; i < 4; ++i) {
    int d = tid + i * 256;
    out[(size_t)row * CD + d] = (vals[i] - mean) * rinv * gamma[d] + beta[d];
  }
}

// xy[row] = concat(x1[row], exog[row]); one block per row
__global__ __launch_bounds__(256) void concat_kernel(
    const float* __restrict__ x1, const float* __restrict__ ex, float* __restrict__ xy)
{
  const int row = blockIdx.x;
  for (int j = threadIdx.x; j < 2 * CD; j += 256) {
    float v = (j < CD) ? x1[(size_t)row * CD + j] : ex[(size_t)row * CD + (j - CD)];
    xy[(size_t)row * 2 * CD + j] = v;
  }
}

// xmean[b,d] = mean over L of x1[b,l,d]; grid (D/256, B)
__global__ __launch_bounds__(256) void mean_seq_kernel(
    const float* __restrict__ x1, float* __restrict__ xm)
{
  const int d = blockIdx.x * 256 + threadIdx.x;
  const int b = blockIdx.y;
  float s = 0.0f;
  for (int l = 0; l < CL; ++l) s += x1[((size_t)b * CL + l) * CD + d];
  xm[(size_t)b * CD + d] = s * (1.0f / CL);
}

// te_rows[row] = dot(h2[row,:256], w3) + b3; one wave per row (8 rows/block)
__global__ __launch_bounds__(256) void te_dot_kernel(
    const float* __restrict__ h2, const float* __restrict__ w3,
    const float* __restrict__ b3, float* __restrict__ rows)
{
  const int lane = threadIdx.x & 31;
  const int wv   = threadIdx.x >> 5;
  const int row  = blockIdx.x * 8 + wv;
  const float* hp = h2 + (size_t)row * CHID;
  float s = 0.0f;
  for (int i = lane; i < CHID; i += 32) s += hp[i] * w3[i];
  for (int m = 16; m > 0; m >>= 1) s += __shfl_xor(s, m);
  if (lane == 0) rows[row] = s + b3[0];
}

// te[b] = mean over L of rows; gate[b] = sigmoid(te[b]); grid B
__global__ __launch_bounds__(256) void te_mean_kernel(
    const float* __restrict__ rows, float* __restrict__ te, float* __restrict__ gate)
{
  __shared__ float red[256];
  const int b = blockIdx.x;
  const int tid = threadIdx.x;
  float s = 0.0f;
  for (int i = tid; i < CL; i += 256) s += rows[(size_t)b * CL + i];
  red[tid] = s; __syncthreads();
  for (int st = 128; st > 0; st >>= 1) { if (tid < st) red[tid] += red[tid + st]; __syncthreads(); }
  if (tid == 0) {
    float m = red[0] * (1.0f / CL);
    te[b] = m;
    gate[b] = sigmoid_f(m);
  }
}

// tau[b] = sigmoid(dot(tau_h[b,:1024], w2) + b2); grid B
__global__ __launch_bounds__(256) void tau_fin_kernel(
    const float* __restrict__ th, const float* __restrict__ w2,
    const float* __restrict__ b2, float* __restrict__ tau)
{
  __shared__ float red[256];
  const int b = blockIdx.x;
  const int tid = threadIdx.x;
  float s = 0.0f;
  #pragma unroll
  for (int i = 0; i < 4; ++i) {
    int d = tid + i * 256;
    s += th[(size_t)b * CD + d] * w2[d];
  }
  red[tid] = s; __syncthreads();
  for (int st = 128; st > 0; st >>= 1) { if (tid < st) red[tid] += red[tid + st]; __syncthreads(); }
  if (tid == 0) tau[b] = sigmoid_f(red[0] + b2[0]);
}

// Write scalar outputs: loss, te[2], tau[2] after the x tensor
__global__ void finalize_kernel(
    const float* __restrict__ te, const float* __restrict__ tau, float* __restrict__ out)
{
  float l = 0.0f;
  for (int b = 0; b < CB; ++b) { float d = 1.0f - te[b]; l += d * d; }
  out[CBLD + 0] = l * (1.0f / CB);
  out[CBLD + 1] = te[0];
  out[CBLD + 2] = te[1];
  out[CBLD + 3] = tau[0];
  out[CBLD + 4] = tau[1];
}

// ---------------------------------------------------------------------------
// Host-side orchestration
// ---------------------------------------------------------------------------
extern "C" void kernel_launch(void* const* d_in, const int* in_sizes, int n_in,
                              void* d_out, int out_size, void* d_ws, size_t ws_size,
                              hipStream_t stream) {
  (void)in_sizes; (void)n_in; (void)out_size; (void)ws_size;

  const float* x      = (const float*)d_in[0];
  const float* exog   = (const float*)d_in[1];
  const float* sa_wq  = (const float*)d_in[2];
  const float* sa_wk  = (const float*)d_in[3];
  const float* sa_wv  = (const float*)d_in[4];
  const float* sa_wo  = (const float*)d_in[5];
  const float* sa_bq  = (const float*)d_in[6];
  const float* sa_bk  = (const float*)d_in[7];
  const float* sa_bv  = (const float*)d_in[8];
  const float* sa_bo  = (const float*)d_in[9];
  const float* ca_wq  = (const float*)d_in[10];
  const float* ca_wk  = (const float*)d_in[11];
  const float* ca_wv  = (const float*)d_in[12];
  const float* ca_wo  = (const float*)d_in[13];
  const float* ca_bq  = (const float*)d_in[14];
  const float* ca_bk  = (const float*)d_in[15];
  const float* ca_bv  = (const float*)d_in[16];
  const float* ca_bo  = (const float*)d_in[17];
  const float* te_w1  = (const float*)d_in[18];
  const float* te_b1  = (const float*)d_in[19];
  const float* te_w2  = (const float*)d_in[20];
  const float* te_b2  = (const float*)d_in[21];
  const float* te_w3  = (const float*)d_in[22];
  const float* te_b3  = (const float*)d_in[23];
  const float* tau_w1 = (const float*)d_in[24];
  const float* tau_b1 = (const float*)d_in[25];
  const float* tau_w2 = (const float*)d_in[26];
  const float* tau_b2 = (const float*)d_in[27];
  const float* ffn_w1 = (const float*)d_in[28];
  const float* ffn_b1 = (const float*)d_in[29];
  const float* ffn_w2 = (const float*)d_in[30];
  const float* ffn_b2 = (const float*)d_in[31];
  const float* ln1_g  = (const float*)d_in[32];
  const float* ln1_b  = (const float*)d_in[33];
  const float* ln2_g  = (const float*)d_in[34];
  const float* ln2_b  = (const float*)d_in[35];
  const float* ln3_g  = (const float*)d_in[36];
  const float* ln3_b  = (const float*)d_in[37];

  float* ws = (float*)d_ws;
  float* qb    = ws;                 // BLD
  float* kb    = ws + 1 * (size_t)CBLD;
  float* vb    = ws + 2 * (size_t)CBLD;
  float* attn  = ws + 3 * (size_t)CBLD;
  float* proj  = ws + 4 * (size_t)CBLD;
  float* x1    = ws + 5 * (size_t)CBLD;
  float* x2    = ws + 6 * (size_t)CBLD;
  float* xy    = ws + 7 * (size_t)CBLD;          // 2*BLD
  float* h1    = ws + 9 * (size_t)CBLD;          // BL*HID
  float* h2    = h1 + (size_t)CBL * CHID;
  float* teR   = h2 + (size_t)CBL * CHID;        // BL
  float* xm    = teR + CBL;                      // B*D
  float* tauh  = xm + CB * CD;                   // B*D
  float* teS   = tauh + CB * CD;                 // B
  float* tauS  = teS + CB;                       // B
  float* gateS = tauS + CB;                      // B
  float* ffnh  = ws;                             // alias q/k/v/attn: 4*BLD floats
  float* outF  = (float*)d_out;

  const dim3 blk256(256), blk128(128);
  const dim3 gProj(CBL / 128, CD / 128);   // M=4096, N=1024
  const dim3 gAttn(CL / 64, CH, CB);
  const dim3 gLN(CBL);

  // ---- Self attention ----
  gemm_bias_kernel<<<gProj, blk256, 0, stream>>>(x, sa_wq, sa_bq, qb, CBL, CD, CD, 0);
  gemm_bias_kernel<<<gProj, blk256, 0, stream>>>(x, sa_wk, sa_bk, kb, CBL, CD, CD, 0);
  gemm_bias_kernel<<<gProj, blk256, 0, stream>>>(x, sa_wv, sa_bv, vb, CBL, CD, CD, 0);
  attn_kernel<<<gAttn, blk128, 0, stream>>>(qb, kb, vb, attn, nullptr, nullptr, CL, CL);
  gemm_bias_kernel<<<gProj, blk256, 0, stream>>>(attn, sa_wo, sa_bo, proj, CBL, CD, CD, 0);
  add_ln_kernel<<<gLN, blk256, 0, stream>>>(x, proj, nullptr, ln1_g, ln1_b, x1, CL);

  // ---- Transfer entropy MLP ----
  concat_kernel<<<gLN, blk256, 0, stream>>>(x1, exog, xy);
  gemm_bias_kernel<<<dim3(CBL / 128, CHID / 128), blk256, 0, stream>>>(xy, te_w1, te_b1, h1, CBL, CHID, 2 * CD, 1);
  gemm_bias_kernel<<<dim3(CBL / 128, CHID / 128), blk256, 0, stream>>>(h1, te_w2, te_b2, h2, CBL, CHID, CHID, 1);
  te_dot_kernel<<<dim3(CBL / 8), blk256, 0, stream>>>(h2, te_w3, te_b3, teR);
  te_mean_kernel<<<dim3(CB), blk256, 0, stream>>>(teR, teS, gateS);

  // ---- Dynamic tau ----
  mean_seq_kernel<<<dim3(CD / 256, CB), blk256, 0, stream>>>(x1, xm);
  gemm_bias_kernel<<<dim3(1, CD / 128), blk256, 0, stream>>>(xm, tau_w1, tau_b1, tauh, CB, CD, CD, 1);
  tau_fin_kernel<<<dim3(CB), blk256, 0, stream>>>(tauh, tau_w2, tau_b2, tauS);

  // ---- Cross attention (te bias, tau temperature, gated residual) ----
  gemm_bias_kernel<<<gProj, blk256, 0, stream>>>(x1,   ca_wq, ca_bq, qb, CBL, CD, CD, 0);
  gemm_bias_kernel<<<gProj, blk256, 0, stream>>>(exog, ca_wk, ca_bk, kb, CBL, CD, CD, 0);
  gemm_bias_kernel<<<gProj, blk256, 0, stream>>>(exog, ca_wv, ca_bv, vb, CBL, CD, CD, 0);
  attn_kernel<<<gAttn, blk128, 0, stream>>>(qb, kb, vb, attn, teS, tauS, CL, CL);
  gemm_bias_kernel<<<gProj, blk256, 0, stream>>>(attn, ca_wo, ca_bo, proj, CBL, CD, CD, 0);
  add_ln_kernel<<<gLN, blk256, 0, stream>>>(x1, proj, gateS, ln2_g, ln2_b, x2, CL);

  // ---- FFN (hidden aliases q/k/v/attn, which are dead now) ----
  gemm_bias_kernel<<<dim3(CBL / 128, CFF / 128), blk256, 0, stream>>>(x2, ffn_w1, ffn_b1, ffnh, CBL, CFF, CD, 1);
  gemm_bias_kernel<<<gProj, blk256, 0, stream>>>(ffnh, ffn_w2, ffn_b2, proj, CBL, CD, CFF, 0);
  add_ln_kernel<<<gLN, blk256, 0, stream>>>(x2, proj, nullptr, ln3_g, ln3_b, outF, CL);

  // ---- Scalars: causality loss, te, tau ----
  finalize_kernel<<<dim3(1), dim3(1), 0, stream>>>(teS, tauS, outF);
}
